// SelfAttention_1640677507101
// MI455X (gfx1250) — matrix operand order repaired and checked by reading, not compiled, not run on hardware
//
#include <hip/hip_runtime.h>
#include <hip/hip_bf16.h>

// ---------------------------------------------------------------------------
// Causal self-attention for S=4096, E=1024, D=1024 (fp32 in/out) on gfx1250.
// Pipeline: fp32->bf16 convert -> 3x WMMA GEMM (Q,K,V; V stored transposed)
//           -> flash-attention kernel (QK^T and P@V on v_wmma_f32_16x16x32_bf16,
//              online softmax in fp32/LDS).
// ---------------------------------------------------------------------------

typedef __bf16 bf16;
typedef __attribute__((ext_vector_type(8)))  __bf16 v8bf;
typedef __attribute__((ext_vector_type(16))) __bf16 v16bf;
typedef __attribute__((ext_vector_type(8)))  float  v8f;

#define S_LEN 4096
#define E_DIM 1024
#define D_DIM 1024

// Build a 16-element bf16 fragment from two 16-byte chunks.
static __device__ __forceinline__ v16bf load_frag(const bf16* p0, const bf16* p1) {
  v8bf lo = *(const v8bf*)p0;
  v8bf hi = *(const v8bf*)p1;
  return __builtin_shufflevector(lo, hi, 0,1,2,3,4,5,6,7,8,9,10,11,12,13,14,15);
}

static __device__ __forceinline__ v8f wmma_bf16(v16bf a, v16bf b, v8f c) {
  // D = A(16x32 bf16) * B(32x16 bf16) + C(16x16 f32)
  return __builtin_amdgcn_wmma_f32_16x16x32_bf16(false, a, false, b, (short)0, c,
                                                 false, false);
}

// ---------------------------------------------------------------------------
// Conversion kernels
// ---------------------------------------------------------------------------
__global__ void cvt_f32_bf16(const float* __restrict__ in, bf16* __restrict__ out,
                             int n) {
  int i = blockIdx.x * blockDim.x + threadIdx.x;
  if (i < n) out[i] = (bf16)in[i];
}

// W (E x D, row-major fp32) -> Wt (D x E, bf16). Contiguous-K layout for WMMA B.
__global__ void cvt_w_transpose(const float* __restrict__ w, bf16* __restrict__ wt) {
  int i = blockIdx.x * blockDim.x + threadIdx.x;   // over E*D
  int e = i / D_DIM;
  int d = i - e * D_DIM;
  wt[(size_t)d * E_DIM + e] = (bf16)w[i];
}

// ---------------------------------------------------------------------------
// Projection GEMM: C(S x D) = A(S x E, bf16) * Bt(D x E, bf16 N-major).
// 256 threads = 8 waves; block tile 128x128; wave tile 64(M) x 32(N) = 8 WMMA accs.
// TRANSPOSE_OUT stores C^T (D x S) -- used for V so P@V B-fragments are contiguous.
// ---------------------------------------------------------------------------
template <bool TRANSPOSE_OUT>
__global__ __launch_bounds__(256)
void gemm_proj(const bf16* __restrict__ A, const bf16* __restrict__ Bt,
               bf16* __restrict__ C) {
  const int lane = threadIdx.x & 31;
  const int wid  = threadIdx.x >> 5;
  const int l16  = lane & 15;
  const int h    = lane >> 4;

  const int ntiles = D_DIM / 128;
  const int bm = blockIdx.x / ntiles;
  const int bn = blockIdx.x % ntiles;
  const int wm = wid >> 2;          // 0..1
  const int wn = wid & 3;           // 0..3
  const int mbase = bm * 128 + wm * 64;
  const int nbase = bn * 128 + wn * 32;

  v8f acc[4][2];
#pragma unroll
  for (int mt = 0; mt < 4; mt++)
#pragma unroll
    for (int nt = 0; nt < 2; nt++)
      acc[mt][nt] = (v8f){0.f, 0.f, 0.f, 0.f, 0.f, 0.f, 0.f, 0.f};

  for (int k0 = 0; k0 < E_DIM; k0 += 32) {
    v16bf afrag[4];
#pragma unroll
    for (int mt = 0; mt < 4; mt++) {
      const bf16* p = A + (size_t)(mbase + mt * 16 + l16) * E_DIM + k0 + 8 * h;
      afrag[mt] = load_frag(p, p + 16);           // K = 8h..8h+7, 16+8h..16+8h+7
    }
    v16bf bfrag[2];
#pragma unroll
    for (int nt = 0; nt < 2; nt++) {
      const bf16* p = Bt + (size_t)(nbase + nt * 16 + l16) * E_DIM + k0 + 8 * h;
      bfrag[nt] = load_frag(p, p + 16);           // K = 8h..8h+7, 16+8h..16+8h+7
    }
#pragma unroll
    for (int mt = 0; mt < 4; mt++)
#pragma unroll
      for (int nt = 0; nt < 2; nt++)
        acc[mt][nt] = wmma_bf16(afrag[mt], bfrag[nt], acc[mt][nt]);
  }

#pragma unroll
  for (int mt = 0; mt < 4; mt++)
#pragma unroll
    for (int nt = 0; nt < 2; nt++)
#pragma unroll
      for (int i = 0; i < 8; i++) {
        const int row = mbase + mt * 16 + i + 8 * h;   // C/D layout: M = vgpr + 8h
        const int col = nbase + nt * 16 + l16;         // N = lane%16
        const bf16 v = (bf16)acc[mt][nt][i];
        if (TRANSPOSE_OUT) C[(size_t)col * S_LEN + row] = v;
        else               C[(size_t)row * D_DIM + col] = v;
      }
}

// ---------------------------------------------------------------------------
// Flash-attention: one block per 16 query rows; 8 waves each own a 128-wide
// D-slice of the output. Key loop in 32-key blocks (causal-truncated).
// ---------------------------------------------------------------------------
__global__ __launch_bounds__(256)
void attn_kernel(const bf16* __restrict__ Q,    // S x D
                 const bf16* __restrict__ K,    // S x D
                 const bf16* __restrict__ Vt,   // D x S (transposed V)
                 float* __restrict__ O) {       // S x D fp32
  __shared__ __align__(16) float sc[16][32];     // score tile (fp32 partial sums)
  __shared__ __align__(16) bf16  pt[16][32];     // P = exp(s - m) as bf16
  __shared__ float m_s[16], l_s[16], corr_s[16];

  const int tid  = threadIdx.x;
  const int lane = tid & 31;
  const int wid  = tid >> 5;
  const int l16  = lane & 15;
  const int h    = lane >> 4;

  const int qb     = blockIdx.x;
  const int qrow0  = qb * 16;
  const int dslice = wid * 128;                  // this wave's output columns

  // Persistent Q fragments for this wave's D-slice (4 x K=32 chunks).
  v16bf qf[4];
#pragma unroll
  for (int c = 0; c < 4; c++) {
    const bf16* p = Q + (size_t)(qrow0 + l16) * D_DIM + dslice + c * 32 + 8 * h;
    qf[c] = load_frag(p, p + 16);
  }

  v8f oacc[8];
#pragma unroll
  for (int nt = 0; nt < 8; nt++)
    oacc[nt] = (v8f){0.f, 0.f, 0.f, 0.f, 0.f, 0.f, 0.f, 0.f};

  if (tid < 16) { m_s[tid] = -1e30f; l_s[tid] = 0.f; }
  __syncthreads();

  const int kb_count = qb / 2 + 1;               // 32-key blocks in causal span
  for (int kb = 0; kb < kb_count; kb++) {
    // Zero the shared score tile.
    for (int i = tid; i < 16 * 32; i += 256) ((float*)sc)[i] = 0.f;
    __syncthreads();

    // Partial QK^T over this wave's D-slice: two 16-key tiles x 4 K-steps.
#pragma unroll
    for (int kt = 0; kt < 2; kt++) {
      v8f s = (v8f){0.f, 0.f, 0.f, 0.f, 0.f, 0.f, 0.f, 0.f};
#pragma unroll
      for (int c = 0; c < 4; c++) {
        // B[d][key] = K[key][d]: contiguous along d in row-major K.
        const bf16* p = K + (size_t)(kb * 32 + kt * 16 + l16) * D_DIM +
                        dslice + c * 32 + 8 * h;
        v16bf kf = load_frag(p, p + 16);
        s = wmma_bf16(qf[c], kf, s);
      }
#pragma unroll
      for (int i = 0; i < 8; i++)
        atomicAdd(&sc[i + 8 * h][kt * 16 + l16], s[i]);   // ds_add_f32 reduce
    }
    __syncthreads();

    // Online softmax (16 rows by threads 0..15), causal mask + 1/sqrt(D)=1/32.
    if (tid < 16) {
      const int r = tid;
      const int qrow = qrow0 + r;
      const float m_old = m_s[r];
      float mx = m_old;
      float v[32];
#pragma unroll
      for (int j = 0; j < 32; j++) {
        float x = sc[r][j] * 0.03125f;
        if (kb * 32 + j > qrow) x = -1e30f;
        v[j] = x;
        mx = fmaxf(mx, x);
      }
      const float corr = __expf(m_old - mx);     // 0 on first block (underflow)
      float sum = 0.f;
#pragma unroll
      for (int j = 0; j < 32; j++) {
        const float p = __expf(v[j] - mx);
        sum += p;
        pt[r][j] = (bf16)p;
      }
      m_s[r] = mx;
      l_s[r] = l_s[r] * corr + sum;
      corr_s[r] = corr;
    }
    __syncthreads();

    // Rescale running output, then O += P(16x32) @ V(32 x 128-slice).
    float cf[8];
#pragma unroll
    for (int i = 0; i < 8; i++) cf[i] = corr_s[i + 8 * h];
#pragma unroll
    for (int nt = 0; nt < 8; nt++)
#pragma unroll
      for (int i = 0; i < 8; i++) oacc[nt][i] *= cf[i];

    const bf16* prow = &pt[l16][0];
    v16bf pf = load_frag(prow + 8 * h, prow + 16 + 8 * h);
#pragma unroll
    for (int nt = 0; nt < 8; nt++) {
      // B[key][col] = Vt[col][key]: contiguous along key in Vt rows.
      const bf16* p = Vt + (size_t)(dslice + nt * 16 + l16) * S_LEN +
                      kb * 32 + 8 * h;
      v16bf vf = load_frag(p, p + 16);
      oacc[nt] = wmma_bf16(pf, vf, oacc[nt]);
    }
    __syncthreads();
  }

  // Normalize by the softmax denominator and store fp32.
  float linv[8];
#pragma unroll
  for (int i = 0; i < 8; i++) linv[i] = 1.0f / l_s[i + 8 * h];
#pragma unroll
  for (int nt = 0; nt < 8; nt++)
#pragma unroll
    for (int i = 0; i < 8; i++) {
      const int row = qrow0 + i + 8 * h;
      const int col = dslice + nt * 16 + l16;
      O[(size_t)row * D_DIM + col] = oacc[nt][i] * linv[i];
    }
}

// ---------------------------------------------------------------------------
// Host launch
// ---------------------------------------------------------------------------
extern "C" void kernel_launch(void* const* d_in, const int* in_sizes, int n_in,
                              void* d_out, int out_size, void* d_ws, size_t ws_size,
                              hipStream_t stream) {
  (void)in_sizes; (void)n_in; (void)out_size; (void)ws_size;

  const float* x  = (const float*)d_in[0];
  const float* Wq = (const float*)d_in[1];
  const float* Wk = (const float*)d_in[2];
  const float* Wv = (const float*)d_in[3];
  float* out = (float*)d_out;

  char* ws = (char*)d_ws;
  bf16* xb  = (bf16*)ws;  ws += (size_t)S_LEN * E_DIM * sizeof(bf16);
  bf16* wt  = (bf16*)ws;  ws += (size_t)E_DIM * D_DIM * sizeof(bf16);  // reused x3
  bf16* Qb  = (bf16*)ws;  ws += (size_t)S_LEN * D_DIM * sizeof(bf16);
  bf16* Kb  = (bf16*)ws;  ws += (size_t)S_LEN * D_DIM * sizeof(bf16);
  bf16* Vtb = (bf16*)ws;  // D x S

  const int nx = S_LEN * E_DIM;
  const int nw = E_DIM * D_DIM;
  cvt_f32_bf16<<<nx / 256, 256, 0, stream>>>(x, xb, nx);

  const dim3 ggrid((S_LEN / 128) * (D_DIM / 128));

  cvt_w_transpose<<<nw / 256, 256, 0, stream>>>(Wq, wt);
  gemm_proj<false><<<ggrid, 256, 0, stream>>>(xb, wt, Qb);

  cvt_w_transpose<<<nw / 256, 256, 0, stream>>>(Wk, wt);
  gemm_proj<false><<<ggrid, 256, 0, stream>>>(xb, wt, Kb);

  cvt_w_transpose<<<nw / 256, 256, 0, stream>>>(Wv, wt);
  gemm_proj<true><<<ggrid, 256, 0, stream>>>(xb, wt, Vtb);

  attn_kernel<<<S_LEN / 16, 256, 0, stream>>>(Qb, Kb, Vtb, out);
}
